// DynamicCurvatureNet_42133629173941
// MI455X (gfx1250) — matrix operand order repaired and checked by reading, not compiled, not run on hardware
//
#include <hip/hip_runtime.h>
#include <cmath>

// ---------------- types for CDNA5 WMMA ----------------
typedef __bf16 bf16_t;
typedef __attribute__((ext_vector_type(16))) __bf16 v16bf;
typedef __attribute__((ext_vector_type(8)))  float  v8f;
typedef __attribute__((ext_vector_type(4)))  unsigned int v4u;

struct BFPair { v4u lo, hi; };

__device__ __forceinline__ v16bf make_frag(v4u lo, v4u hi) {
    BFPair p{lo, hi};
    return __builtin_bit_cast(v16bf, p);
}

// XOR-lane swap via ds_swizzle_b32, mask fully in the immediate:
// group-of-32 encoding offset[15]=0: and_mask[4:0]=0x1F, or_mask[9:5]=0, xor_mask[14:10]=S.
template <int S>
__device__ __forceinline__ float swz_xor(float v) {
    int i = __builtin_bit_cast(int, v);
    i = __builtin_amdgcn_ds_swizzle(i, (S << 10) | 0x1F);
    return __builtin_bit_cast(float, i);
}

// A-matrix 16x32 bf16 fragment from LDS (row-major, stride in halves).
// ISA layout: lanes 0-15 (M=lane): K 0..7 then 16..23; lanes 16-31 (M=lane-16): K 8..15 then 24..31.
__device__ __forceinline__ v16bf load_fragA(const bf16_t* base, int stride,
                                            int row, int kbase, int lid) {
    const bf16_t* p = base + row * stride + kbase + ((lid < 16) ? 0 : 8);
    v4u lo = *(const v4u*)p;          // 8 halves
    v4u hi = *(const v4u*)(p + 16);   // 8 halves
    return make_frag(lo, hi);
}

// B-matrix 32x16 bf16 fragment built from global fp32 weights, W row-major [N][K].
// ISA layout: lanes 0-15 (N=lane): K 0..7 then 16..23; lanes 16-31 (N=lane-16): K 8..15 then 24..31.
__device__ __forceinline__ v16bf load_fragB_f32(const float* __restrict__ wrow,
                                                int kbase, int lid) {
    const float* p = wrow + kbase + ((lid < 16) ? 0 : 8);
    float4 a = ((const float4*)p)[0];
    float4 b = ((const float4*)p)[1];
    float4 c = ((const float4*)p)[4];
    float4 d = ((const float4*)p)[5];
    v16bf r;
    r[0]=(bf16_t)a.x;  r[1]=(bf16_t)a.y;  r[2]=(bf16_t)a.z;  r[3]=(bf16_t)a.w;
    r[4]=(bf16_t)b.x;  r[5]=(bf16_t)b.y;  r[6]=(bf16_t)b.z;  r[7]=(bf16_t)b.w;
    r[8]=(bf16_t)c.x;  r[9]=(bf16_t)c.y;  r[10]=(bf16_t)c.z; r[11]=(bf16_t)c.w;
    r[12]=(bf16_t)d.x; r[13]=(bf16_t)d.y; r[14]=(bf16_t)d.z; r[15]=(bf16_t)d.w;
    return r;
}

// ---------------- kernel 1: SST gate, per-block partial sums ----------------
// Each block handles 256 rows (4 tiles of 64), writes one deterministic partial sum.
__global__ __launch_bounds__(256)
void dyn_gate_kernel(const float* __restrict__ x,
                     const float* __restrict__ w1,   // [8][128]
                     const float* __restrict__ b1,   // [8]
                     const float* __restrict__ w2,   // [1][8]
                     const float* __restrict__ b2,   // [1]
                     float* __restrict__ partials) {
    extern __shared__ char smem[];
    float* w1s = (float*)smem;                  // 1024 f  (4096 B)
    float* xs  = (float*)(smem + 4096);         // 64*132  (33792 B)
    float* red = (float*)(smem + 4096 + 33792); // 256 f

    const int t = threadIdx.x;

    {   // stage w1 (8x128) linear [j*128+k]
        const float4* wv = (const float4*)w1;
        float4 v = wv[t];                       // 256 float4 exactly
        w1s[t*4+0]=v.x; w1s[t*4+1]=v.y; w1s[t*4+2]=v.z; w1s[t*4+3]=v.w;
    }

    float sacc = 0.f;
    for (int it = 0; it < 4; ++it) {
        __syncthreads();    // protect xs reuse across iterations
        const long rbase = (long)blockIdx.x * 256 + it * 64;
        // stage 64x128 fp32 rows, padded stride 132
        const float4* xv = (const float4*)(x + rbase * 128);
        #pragma unroll
        for (int i = 0; i < 8; ++i) {
            int idx = t + i * 256;              // 2048 float4
            float4 v = xv[idx];
            int row = idx >> 5;
            int col = (idx & 31) << 2;
            float* d = xs + row * 132 + col;
            d[0]=v.x; d[1]=v.y; d[2]=v.z; d[3]=v.w;
        }
        __syncthreads();
        if (t < 64) {
            float h[8];
            #pragma unroll
            for (int j = 0; j < 8; ++j) h[j] = 0.f;
            const float* xr = xs + t * 132;
            for (int k = 0; k < 128; ++k) {
                float xv_ = xr[k];
                #pragma unroll
                for (int j = 0; j < 8; ++j) h[j] += xv_ * w1s[j * 128 + k];
            }
            float sdot = b2[0];
            #pragma unroll
            for (int j = 0; j < 8; ++j) {
                float hj = 1.f / (1.f + expf(-(h[j] + b1[j])));
                sdot += w2[j] * hj;
            }
            sacc += 1.f / (1.f + expf(-sdot));
        }
    }
    __syncthreads();
    red[t] = sacc;                              // zero for t>=64
    for (int off = 128; off > 0; off >>= 1) {
        __syncthreads();
        if (t < off) red[t] += red[t + off];
    }
    __syncthreads();
    if (t == 0) partials[blockIdx.x] = red[0];
}

// ---------------- kernel 2: finalize scalars (1 block, deterministic) ----------------
__global__ __launch_bounds__(256)
void dyn_finalize_kernel(const float* __restrict__ partials, int np, long Btot,
                         const float* __restrict__ gvip,
                         const float* __restrict__ bias,   // [128]
                         const float* __restrict__ clsw,   // [32][128]
                         float* __restrict__ wsf,          // header: c, sc, b2sum, bc[32]
                         float* __restrict__ gammaOut) {
    __shared__ float red[256];
    const int t = threadIdx.x;
    float s = 0.f;
    for (int i = t; i < np; i += 256) s += partials[i];   // fixed order
    red[t] = s;
    for (int off = 128; off > 0; off >>= 1) {
        __syncthreads();
        if (t < off) red[t] += red[t + off];
    }
    __syncthreads();
    if (t == 0) {
        float lg = red[0] / (float)Btot;
        float g = fminf(fmaxf(lg + gvip[0], 0.f), 1.f);
        float c = fminf(fmaxf(g * 5.f, 1e-4f), 5.f);
        wsf[0] = c;
        wsf[1] = sqrtf(c);
        *gammaOut = g;
    }
    __syncthreads();
    float bsq = 0.f;
    if (t < 128) { float b = bias[t]; bsq = b * b; }
    red[t] = bsq;
    for (int off = 128; off > 0; off >>= 1) {
        __syncthreads();
        if (t < off) red[t] += red[t + off];
    }
    __syncthreads();
    if (t == 0) wsf[2] = red[0];
    if (t < 32) {
        float acc = 0.f;
        for (int k = 0; k < 128; ++k) acc += bias[k] * clsw[t * 128 + k];
        wsf[3 + t] = acc;                                 // bc[o] = bias . cls_w[o]
    }
}

// ---------------- kernel 3: fused hyperbolic layer (WMMA bf16) ----------------
// Block = 256 threads (8 wave32), 64 rows per block.
// LDS layout (bytes):
#define XB_OFF    0            // bf16 x tile   64*136*2 = 17408
#define GB_OFF    17408        // bf16 g tile   64*136*2 = 17408
#define BIAS_OFF  34816        // f32 bias      128*4
#define CLSB_OFF  35328        // f32 cls_b     32*4
#define BC_OFF    35456        // f32 bc        32*4
#define PX_OFF    35584        // f32 xsq part  64*4*4
#define PSQ_OFF   36608        // f32 gsq part  64*8*4
#define PGB_OFF   38656        // f32 g.b part  64*8*4
#define RP_OFF    40704        // f32 rowp      64*4
#define RQ_OFF    40960        // f32 rowq      64*4
#define SMEM3     41216
#define WSTR      136          // padded bf16 row stride (272B, 16B-aligned, bank-staggered)

__global__ __launch_bounds__(256)
void dyn_hyp_main_kernel(const float* __restrict__ x,     // [B][128]
                         const float* __restrict__ W,     // [128][128] hyp_weight
                         const float* __restrict__ bias,  // [128]
                         const float* __restrict__ clsw,  // [32][128]
                         const float* __restrict__ clsb,  // [32]
                         const float* __restrict__ wsf,   // c, sc, b2sum, bc[32]
                         float* __restrict__ out) {       // [B][32]
    extern __shared__ char smem[];
    bf16_t* xb    = (bf16_t*)(smem + XB_OFF);
    bf16_t* gb16  = (bf16_t*)(smem + GB_OFF);
    float*  biasS = (float*)(smem + BIAS_OFF);
    float*  clsbS = (float*)(smem + CLSB_OFF);
    float*  bcS   = (float*)(smem + BC_OFF);
    float*  partx = (float*)(smem + PX_OFF);
    float*  pgsq  = (float*)(smem + PSQ_OFF);
    float*  pgb   = (float*)(smem + PGB_OFF);
    float*  rowp  = (float*)(smem + RP_OFF);
    float*  rowq  = (float*)(smem + RQ_OFF);

    const int t   = threadIdx.x;
    const int lid = t & 31;
    const int wid = t >> 5;
    const long row0 = (long)blockIdx.x * 64;

    // ---- stage x tile (fp32 -> bf16) + xsq partials ----
    {
        int row = t >> 2, seg = t & 3;                    // 64 rows x 4 segs of 32
        const float4* xp = (const float4*)(x + (row0 + row) * 128 + seg * 32);
        bf16_t* d = xb + row * WSTR + seg * 32;
        float sq = 0.f;
        #pragma unroll
        for (int i = 0; i < 8; ++i) {
            float4 v = xp[i];
            sq += v.x*v.x + v.y*v.y + v.z*v.z + v.w*v.w;
            d[i*4+0]=(bf16_t)v.x; d[i*4+1]=(bf16_t)v.y;
            d[i*4+2]=(bf16_t)v.z; d[i*4+3]=(bf16_t)v.w;
        }
        partx[row * 4 + seg] = sq;
    }
    if (t < 128) biasS[t] = bias[t];
    if (t < 32)  { clsbS[t] = clsb[t]; bcS[t] = wsf[3 + t]; }
    __syncthreads();

    // ---- GEMM1: g = x @ W^T, wave `wid` owns output columns wid*16..wid*16+15 ----
    {
        const int n = wid;
        const int nn = n * 16 + (lid & 15);               // output column of this lane
        v16bf bfr[4];
        #pragma unroll
        for (int kk = 0; kk < 4; ++kk)
            bfr[kk] = load_fragB_f32(W + nn * 128, kk * 32, lid);   // B[k][n] = W[n][k]
        const float bcol = biasS[nn];

        #pragma unroll
        for (int m = 0; m < 4; ++m) {
            v8f acc;
            #pragma unroll
            for (int i = 0; i < 8; ++i) acc[i] = 0.f;
            #pragma unroll
            for (int kk = 0; kk < 4; ++kk) {
                v16bf af = load_fragA(xb, WSTR, m * 16 + (lid & 15), kk * 32, lid);
                acc = __builtin_amdgcn_wmma_f32_16x16x32_bf16(
                          false, af, false, bfr[kk], (short)0, acc, false, false);
            }
            // C layout: lanes 0-15 rows m*16+r, lanes 16-31 rows m*16+8+r, col = nn.
            const int rb = m * 16 + ((lid < 16) ? 0 : 8);
            #pragma unroll
            for (int r = 0; r < 8; ++r) {
                float v = acc[r];
                gb16[(rb + r) * WSTR + nn] = (bf16_t)v;   // bf16 copy for GEMM2
                // 16-lane XOR-tree reduction of g^2 and g*bias over this wave's 16 cols
                float sq = v * v;
                float bd = v * bcol;
                sq += swz_xor<1>(sq);  bd += swz_xor<1>(bd);
                sq += swz_xor<2>(sq);  bd += swz_xor<2>(bd);
                sq += swz_xor<4>(sq);  bd += swz_xor<4>(bd);
                sq += swz_xor<8>(sq);  bd += swz_xor<8>(bd);
                if ((lid & 15) == 0) {
                    pgsq[(rb + r) * 8 + n] = sq;
                    pgb [(rb + r) * 8 + n] = bd;
                }
            }
        }
    }
    __syncthreads();

    // ---- per-row scalar chain (expmap0 / mobius_matvec / mobius_add / logmap0) ----
    if (t < 64) {
        const float c  = wsf[0];
        const float sc = wsf[1];
        const float b2 = wsf[2];
        float xsq = partx[t*4] + partx[t*4+1] + partx[t*4+2] + partx[t*4+3];
        float gsq = 0.f, gbd = 0.f;
        #pragma unroll
        for (int i = 0; i < 8; ++i) { gsq += pgsq[t*8+i]; gbd += pgb[t*8+i]; }

        const float EPS   = 1e-15f;
        const float ACLIP = 1.f - 1e-7f;
        const float MAXT  = 1.f - 4e-3f;                  // geoopt ball eps

        float xn  = fmaxf(sqrtf(xsq), EPS);
        float a   = sc * xn;
        float th  = tanhf(a);
        float f1  = ((th > MAXT) ? MAXT : th) / a;        // x_hyp = f1 * x (incl. project)
        float xnh = fmaxf(f1 * xn, EPS);
        float gnr = sqrtf(gsq);
        float mxn = fmaxf(f1 * gnr, EPS);
        float z   = fminf(sc * xnh, ACLIP);
        float ar  = 0.5f * logf((1.f + z) / (1.f - z));   // artanh
        float tt  = tanhf(mxn / xnh * ar);
        float sres = (gnr > 0.f) ? (tt / (mxn * sc)) * f1 : 0.f;   // res = sres * g
        float resn = sres * gnr;
        float maxn = MAXT / sc;
        if (resn > maxn) { sres *= maxn / resn; resn = maxn; }
        // mobius_add(res, bias)
        float x2  = resn * resn;
        float xy  = sres * gbd;
        float k1  = 1.f + 2.f * c * xy + c * b2;
        float k2  = 1.f - c * x2;
        float den = fmaxf(1.f + 2.f * c * xy + c * c * x2 * b2, EPS);
        float u   = (k1 / den) * sres;                    // out_hyp = u*g + v*bias
        float v   =  k2 / den;
        float on  = sqrtf(fmaxf(u*u*gsq + 2.f*u*v*gbd + v*v*b2, 0.f));
        if (on > maxn) { float sp = maxn / on; u *= sp; v *= sp; on = maxn; }
        float onc = fmaxf(on, EPS);
        float z2  = fminf(sc * onc, ACLIP);
        float ar2 = 0.5f * logf((1.f + z2) / (1.f - z2));
        float hs  = ar2 / (sc * onc);                     // hidden = hs * out_hyp
        rowp[t] = hs * u;
        rowq[t] = hs * v;
    }
    __syncthreads();

    // ---- GEMM2: out = p*(g @ cls_w^T) + q*bc + cls_b ; 8 tiles, one per wave ----
    {
        const int m2 = wid >> 1;                          // row tile 0..3
        const int n2 = wid & 1;                           // col tile 0..1
        const int nn = n2 * 16 + (lid & 15);              // output column (0..31)
        v8f acc;
        #pragma unroll
        for (int i = 0; i < 8; ++i) acc[i] = 0.f;
        #pragma unroll
        for (int kk = 0; kk < 4; ++kk) {
            v16bf af  = load_fragA(gb16, WSTR, m2 * 16 + (lid & 15), kk * 32, lid);
            v16bf bfr = load_fragB_f32(clsw + nn * 128, kk * 32, lid);
            acc = __builtin_amdgcn_wmma_f32_16x16x32_bf16(
                      false, af, false, bfr, (short)0, acc, false, false);
        }
        const int rb = m2 * 16 + ((lid < 16) ? 0 : 8);
        const float bcq = bcS[nn];
        const float cb  = clsbS[nn];
        #pragma unroll
        for (int r = 0; r < 8; ++r) {
            const int row = rb + r;
            out[(row0 + row) * 32 + nn] = rowp[row] * acc[r] + rowq[row] * bcq + cb;
        }
    }
}

// ---------------- host launcher ----------------
extern "C" void kernel_launch(void* const* d_in, const int* in_sizes, int n_in,
                              void* d_out, int out_size, void* d_ws, size_t ws_size,
                              hipStream_t stream) {
    const float* x    = (const float*)d_in[0];
    const float* gvip = (const float*)d_in[1];
    const float* w1   = (const float*)d_in[2];
    const float* b1   = (const float*)d_in[3];
    const float* w2   = (const float*)d_in[4];
    const float* b2   = (const float*)d_in[5];
    const float* W    = (const float*)d_in[6];
    const float* bias = (const float*)d_in[7];
    const float* clsw = (const float*)d_in[8];
    const float* clsb = (const float*)d_in[9];
    float* out = (float*)d_out;

    const long Btot = (long)in_sizes[0] / 128;            // 1048576
    float* wsf      = (float*)d_ws;                       // [0]=c [1]=sc [2]=b2sum [3..34]=bc
    float* partials = wsf + 64;

    const int nbGate = (int)(Btot / 256);                 // 4096 partials
    dyn_gate_kernel<<<nbGate, 256, 4096 + 33792 + 1024, stream>>>(
        x, w1, b1, w2, b2, partials);

    dyn_finalize_kernel<<<1, 256, 0, stream>>>(
        partials, nbGate, Btot, gvip, bias, clsw, wsf, out + Btot * 32);

    const int nbMain = (int)(Btot / 64);                  // 16384 blocks
    dyn_hyp_main_kernel<<<nbMain, 256, SMEM3, stream>>>(
        x, W, bias, clsw, clsb, wsf, out);
}
